// AttentionBlock_6201932775590
// MI455X (gfx1250) — matrix operand order repaired and checked by reading, not compiled, not run on hardware
//
#include <hip/hip_runtime.h>
#include <hip/hip_bf16.h>

typedef __attribute__((ext_vector_type(16))) _Float16 v16h;
typedef __attribute__((ext_vector_type(8)))  _Float16 v8h;
typedef __attribute__((ext_vector_type(8)))  float    v8f;

#define BATCH   32
#define CIMG    224
#define CIN     256
#define KD      256
#define S_TOT   1024

static __device__ __forceinline__ v8f zero8() {
    v8f z;
#pragma unroll
    for (int i = 0; i < 8; ++i) z[i] = 0.0f;
    return z;
}

// A fragment: 16x32 (MxK), f16, row-major source with row stride ld (halfs).
// Lanes 0-15 row M=lane hold K=0..7 / K=16..23; lanes 16-31 hold K=8..15 / K=24..31.
static __device__ __forceinline__ v16h load_a_frag(const _Float16* src, int ld) {
    int lane = threadIdx.x & 31;
    int row  = lane & 15;
    int kb   = (lane >> 4) << 3;          // 0 or 8
    const _Float16* p = src + (size_t)row * ld + kb;
    v8h lo = *(const v8h*)(p);            // K = kb .. kb+7
    v8h hi = *(const v8h*)(p + 16);       // K = 16+kb .. 16+kb+7
    v16h a;
#pragma unroll
    for (int i = 0; i < 8; ++i) { a[i] = lo[i]; a[8 + i] = hi[i]; }
    return a;
}

// B fragment: 32x16 (KxN), f16, supplied as its N-major transpose:
// srcT[n*ld + k] == B[k][n].  Lane n holds column n, K rows kb..kb+7 / 16+kb..16+kb+7.
static __device__ __forceinline__ v16h load_b_frag_nt(const _Float16* srcT, int ld) {
    int lane = threadIdx.x & 31;
    int n    = lane & 15;
    int kb   = (lane >> 4) << 3;          // 0 or 8
    const _Float16* p = srcT + (size_t)n * ld + kb;
    v8h lo = *(const v8h*)(p);
    v8h hi = *(const v8h*)(p + 16);
    v16h r;
#pragma unroll
    for (int i = 0; i < 8; ++i) { r[i] = lo[i]; r[8 + i] = hi[i]; }
    return r;
}

static __device__ __forceinline__ v8f wmma_f16(v16h a, v16h b, v8f c) {
    return __builtin_amdgcn_wmma_f32_16x16x32_f16(false, a, false, b,
                                                  (short)0, c, false, false);
}

// ---------------------------------------------------------------------------
// Kernel 1: build x0[b][s][c] (f16) = concat(x, positional enc), transposed.
// ---------------------------------------------------------------------------
__global__ void build_x0_kernel(const float* __restrict__ x, _Float16* __restrict__ x0) {
    int idx = blockIdx.x * blockDim.x + threadIdx.x;
    if (idx >= BATCH * S_TOT * CIN) return;
    int c = idx & (CIN - 1);
    int s = (idx >> 8) & (S_TOT - 1);
    int b = idx >> 18;
    float val;
    if (c < CIMG) {
        val = x[((size_t)b * CIMG + c) * S_TOT + s];
    } else {
        int ch = c - CIMG;                 // 0..31
        int h = s >> 5, w = s & 31;
        int i = (ch < 16) ? (ch * 32 + h) : ((ch - 16) * 32 + w);  // raw-reshape view
        int pos = i >> 4, j = i & 15;
        if (pos == 0) {
            val = 0.0f;
        } else {
            float ang = (float)pos * __powf(10000.0f, -(float)(2 * (j >> 1)) * (1.0f / 16.0f));
            val = (j & 1) ? __cosf(ang) : __sinf(ang);
        }
    }
    x0[idx] = (_Float16)val;
}

// ---------------------------------------------------------------------------
// Kernel 2: convert wq|wk|wv to f16, packed [3][256][256]
// ---------------------------------------------------------------------------
__global__ void cvt_w_kernel(const float* __restrict__ wq, const float* __restrict__ wk,
                             const float* __restrict__ wv, _Float16* __restrict__ Wf) {
    int idx = blockIdx.x * blockDim.x + threadIdx.x;
    if (idx >= 3 * KD * CIN) return;
    const float* src = (idx < KD * CIN) ? wq : ((idx < 2 * KD * CIN) ? wk : wv);
    Wf[idx] = (_Float16)src[idx & (KD * CIN - 1)];
}

// ---------------------------------------------------------------------------
// Kernel 3: QKV projection.  q = relu(W @ x0 + b).
// grid = (B*16, 3), block = 256 (8 waves).  Wave owns a 16(s) x 128(k) tile.
// Q,K stored [b][s][k]; V stored [b][v][t].
// Fragment loads are software-pipelined one step ahead of the WMMAs.
// ---------------------------------------------------------------------------
__global__ __launch_bounds__(256)
void qkv_proj_kernel(const _Float16* __restrict__ x0, const _Float16* __restrict__ Wf,
                     const float* __restrict__ bq, const float* __restrict__ bk,
                     const float* __restrict__ bv,
                     _Float16* __restrict__ Qo, _Float16* __restrict__ Ko,
                     _Float16* __restrict__ Vo) {
    int blk   = blockIdx.x;
    int which = blockIdx.y;                 // 0=Q 1=K 2=V
    int b     = blk >> 4;
    int sblk  = blk & 15;
    int wave  = threadIdx.x >> 5;
    int lane  = threadIdx.x & 31;
    int sbase = sblk * 64 + (wave & 3) * 16;
    int k128  = (wave >> 2) * 128;

    const _Float16* W    = Wf + (size_t)which * KD * CIN + (size_t)k128 * CIN;
    const float*    bias = (which == 0) ? bq : ((which == 1) ? bk : bv);
    const _Float16* xa   = x0 + ((size_t)b * S_TOT + sbase) * CIN;

    v8f acc[8];
#pragma unroll
    for (int j = 0; j < 8; ++j) acc[j] = zero8();

    v16h a_cur = load_a_frag(xa, CIN);
    v16h b_cur = load_b_frag_nt(W, CIN);
    v16h a_nxt = a_cur;
#pragma unroll
    for (int i = 0; i < 64; ++i) {          // i = c*8 + j
        int j = i & 7;
        v16h b_nxt = b_cur;
        if (i < 63) {
            int c2 = (i + 1) >> 3, j2 = (i + 1) & 7;
            b_nxt = load_b_frag_nt(W + (size_t)(j2 * 16) * CIN + c2 * 32, CIN);
            if (j2 == 0) a_nxt = load_a_frag(xa + c2 * 32, CIN);
        }
        acc[j] = wmma_f16(a_cur, b_cur, acc[j]);
        b_cur = b_nxt;
        if (j == 7) a_cur = a_nxt;
    }

    int ncol  = lane & 15;
    int half8 = (lane >> 4) << 3;
#pragma unroll
    for (int j = 0; j < 8; ++j) {
        int   kg = k128 + j * 16 + ncol;
        float bb = bias[kg];
#pragma unroll
        for (int r = 0; r < 8; ++r) {
            float    v = fmaxf(acc[j][r] + bb, 0.0f);
            _Float16 h = (_Float16)v;
            int      sg = sbase + half8 + r;
            if (which == 0)      Qo[((size_t)b * S_TOT + sg) * KD + kg] = h;
            else if (which == 1) Ko[((size_t)b * S_TOT + sg) * KD + kg] = h;
            else                 Vo[((size_t)b * KD + kg) * S_TOT + sg] = h;
        }
    }
}

// ---------------------------------------------------------------------------
// Kernel 4: causal flash attention.  grid = (B, 8), block = 256 (8 waves).
// Each wave owns 16 query rows, iterates t-chunks of 32, online softmax,
// O[16s x 256v] accumulated in 16 f32 fragments, epilogue writes [b][v][s].
// ---------------------------------------------------------------------------
__global__ __launch_bounds__(256)
void attn_flash_kernel(const _Float16* __restrict__ Q,   // [b][s][kd]
                       const _Float16* __restrict__ Kt,  // [b][t][kd]
                       const _Float16* __restrict__ Vt,  // [b][v][t]
                       float* __restrict__ out) {        // [b][v][s]
    __shared__ _Float16 ldsP[8][16 * 32];   // per-wave 16x32 P tile (C->A relayout)

    int b     = blockIdx.x;
    int wave  = threadIdx.x >> 5;
    int lane  = threadIdx.x & 31;
    int stile = blockIdx.y * 8 + wave;
    int sbase = stile * 16;

    const _Float16* Qb = Q  + (size_t)b * S_TOT * KD;
    const _Float16* Kb = Kt + (size_t)b * S_TOT * KD;
    const _Float16* Vb = Vt + (size_t)b * KD * S_TOT;

    // Preload the wave's 16x256 Q block as 8 A-fragments.
    v16h qa[8];
#pragma unroll
    for (int c = 0; c < 8; ++c)
        qa[c] = load_a_frag(Qb + (size_t)sbase * KD + c * 32, KD);

    v8f o[16];
#pragma unroll
    for (int j = 0; j < 16; ++j) o[j] = zero8();
    float mrow[8], lrow[8];
#pragma unroll
    for (int r = 0; r < 8; ++r) { mrow[r] = -3.0e38f; lrow[r] = 0.0f; }

    int ncol  = lane & 15;
    int half8 = (lane >> 4) << 3;
    int nchunk = (sbase + 16 + 31) >> 5;

    for (int ch = 0; ch < nchunk; ++ch) {
        int tbase = ch << 5;

        // prefetch next chunk: K rows (32 x 512B) and V columns (256 x 64B)
        if (ch + 1 < nchunk) {
            const _Float16* pk = Kb + (size_t)(tbase + 32 + lane) * KD;
            __builtin_prefetch(pk, 0, 1);
            __builtin_prefetch(pk + 128, 0, 1);
#pragma unroll
            for (int q = 0; q < 8; ++q)
                __builtin_prefetch(Vb + (size_t)(q * 32 + lane) * S_TOT + tbase + 32, 0, 1);
        }

        // ---- scores: S[16 x 32] = Q(16x256) @ K^T, two 16-col tiles ----
        v8f s0 = zero8(), s1 = zero8();
        v16h kb0 = load_b_frag_nt(Kb + (size_t)tbase * KD, KD);
        v16h kb1 = load_b_frag_nt(Kb + (size_t)(tbase + 16) * KD, KD);
#pragma unroll
        for (int c = 0; c < 8; ++c) {
            v16h n0 = kb0, n1 = kb1;
            if (c < 7) {
                n0 = load_b_frag_nt(Kb + (size_t)tbase        * KD + (c + 1) * 32, KD);
                n1 = load_b_frag_nt(Kb + (size_t)(tbase + 16) * KD + (c + 1) * 32, KD);
            }
            s0 = wmma_f16(qa[c], kb0, s0);
            s1 = wmma_f16(qa[c], kb1, s1);
            kb0 = n0; kb1 = n1;
        }
#pragma unroll
        for (int r = 0; r < 8; ++r) { s0[r] *= 0.0625f; s1[r] *= 0.0625f; }

        if (tbase + 31 > sbase) {          // causal mask only near the diagonal
#pragma unroll
            for (int r = 0; r < 8; ++r) {
                int srow = sbase + half8 + r;
                if (tbase + ncol      > srow) s0[r] = -1.0e30f;
                if (tbase + 16 + ncol > srow) s1[r] = -1.0e30f;
            }
        }

        // ---- online softmax (row stats via 16-lane xor-shuffle reduce) ----
#pragma unroll
        for (int r = 0; r < 8; ++r) {
            float mx = fmaxf(s0[r], s1[r]);
#pragma unroll
            for (int off = 8; off >= 1; off >>= 1)
                mx = fmaxf(mx, __shfl_xor(mx, off, 32));
            float mnew = fmaxf(mrow[r], mx);
            float corr = __expf(mrow[r] - mnew);
            float p0 = __expf(s0[r] - mnew);
            float p1 = __expf(s1[r] - mnew);
            float sm = p0 + p1;
#pragma unroll
            for (int off = 8; off >= 1; off >>= 1)
                sm += __shfl_xor(sm, off, 32);
            lrow[r] = lrow[r] * corr + sm;
            mrow[r] = mnew;
#pragma unroll
            for (int j = 0; j < 16; ++j) o[j][r] *= corr;
            // stash P (C-layout element -> row-major LDS tile)
            ldsP[wave][(half8 + r) * 32 + ncol]      = (_Float16)p0;
            ldsP[wave][(half8 + r) * 32 + 16 + ncol] = (_Float16)p1;
        }

        // ---- O += P(16x32) @ V(32x256), V-frags pipelined one ahead ----
        v16h pa = load_a_frag(&ldsP[wave][0], 32);
        v16h vb = load_b_frag_nt(Vb + tbase, S_TOT);
#pragma unroll
        for (int j = 0; j < 16; ++j) {
            v16h vn = vb;
            if (j < 15)
                vn = load_b_frag_nt(Vb + (size_t)((j + 1) * 16) * S_TOT + tbase, S_TOT);
            o[j] = wmma_f16(pa, vb, o[j]);
            vb = vn;
        }
    }

    // ---- epilogue: multiply by 1/denom (v_rcp_f32), write out[b][v][s] ----
    float rinv[8];
#pragma unroll
    for (int r = 0; r < 8; ++r) rinv[r] = __builtin_amdgcn_rcpf(lrow[r]);

    float* ob = out + (size_t)b * KD * S_TOT;
#pragma unroll
    for (int j = 0; j < 16; ++j) {
#pragma unroll
        for (int r = 0; r < 8; ++r) {
            float val = o[j][r] * rinv[r];
            ob[(size_t)(j * 16 + ncol) * S_TOT + sbase + half8 + r] = val;
        }
    }
}

// ---------------------------------------------------------------------------
extern "C" void kernel_launch(void* const* d_in, const int* in_sizes, int n_in,
                              void* d_out, int out_size, void* d_ws, size_t ws_size,
                              hipStream_t stream) {
    const float* x  = (const float*)d_in[0];
    const float* wq = (const float*)d_in[1];
    const float* bq = (const float*)d_in[2];
    const float* wk = (const float*)d_in[3];
    const float* bk = (const float*)d_in[4];
    const float* wv = (const float*)d_in[5];
    const float* bv = (const float*)d_in[6];
    float* out = (float*)d_out;

    // workspace carve-up (halfs)
    _Float16* x0 = (_Float16*)d_ws;                       // 32*1024*256
    _Float16* Wf = x0 + (size_t)BATCH * S_TOT * CIN;      // 3*256*256
    _Float16* Qb = Wf + 3 * KD * CIN;                     // 32*1024*256
    _Float16* Kb = Qb + (size_t)BATCH * S_TOT * KD;       // 32*1024*256
    _Float16* Vb = Kb + (size_t)BATCH * S_TOT * KD;       // 32*256*1024

    {
        int n = BATCH * S_TOT * CIN;
        build_x0_kernel<<<(n + 255) / 256, 256, 0, stream>>>(x, x0);
    }
    {
        int n = 3 * KD * CIN;
        cvt_w_kernel<<<(n + 255) / 256, 256, 0, stream>>>(wq, wk, wv, Wf);
    }
    qkv_proj_kernel<<<dim3(BATCH * 16, 3), 256, 0, stream>>>(x0, Wf, bq, bk, bv, Qb, Kb, Vb);
    attn_flash_kernel<<<dim3(BATCH, 8), 256, 0, stream>>>(Qb, Kb, Vb, out);
}
